// ModernBertAttention_21500606284278
// MI455X (gfx1250) — matrix operand order repaired and checked by reading, not compiled, not run on hardware
//
#include <hip/hip_runtime.h>
#include <math.h>

typedef unsigned short u16;
typedef unsigned int   u32;
typedef __attribute__((ext_vector_type(2)))  float  v2f;
typedef __attribute__((ext_vector_type(8)))  float  v8f;
typedef __attribute__((ext_vector_type(16))) __bf16 v16bf;

#define BSZ   4
#define SEQ   2048
#define HID   768
#define NH    12
#define HD    64
#define NQKV  2304          // 3*HID
#define MTOT  (BSZ*SEQ)     // 8192
#define HALF_WIN 64
#define NEGBIG (-3.0e38f)
#define KCHUNKS (HID/32)    // 24

// ---------------- WMMA wrappers ----------------
__device__ __forceinline__ v8f wmma4(v2f a, v2f b, v8f c) {
  return __builtin_amdgcn_wmma_f32_16x16x4_f32(false, a, false, b, (short)0, c, false, false);
}
__device__ __forceinline__ v8f wmma32bf(v16bf a, v16bf b, v8f c) {
  return __builtin_amdgcn_wmma_f32_16x16x32_bf16(false, a, false, b, (short)0, c, false, false);
}

// ---------------- bf16 split helpers (RNE) ----------------
__device__ __forceinline__ u16 f2bf(float f) {
  u32 u = __float_as_uint(f);
  u = (u + 0x7fffu + ((u >> 16) & 1u)) >> 16;
  return (u16)u;
}
__device__ __forceinline__ float bf2f(u16 h) {
  return __uint_as_float(((u32)h) << 16);
}
__device__ __forceinline__ int clampi(int x, int lo, int hi) {
  return x < lo ? lo : (x > hi ? hi : x);
}

struct B32 { uint4 lo, hi; };
__device__ __forceinline__ v16bf mk16(uint4 a, uint4 b) {
  B32 t{a, b};
  return __builtin_bit_cast(v16bf, t);
}

// ---------------- CDNA5 async copy: global -> LDS, 16B per lane -------------
__device__ __forceinline__ void async_b128(u32 lds_off, const void* gaddr) {
  asm volatile("global_load_async_to_lds_b128 %0, %1, off"
               :: "v"(lds_off), "v"(gaddr) : "memory");
}
__device__ __forceinline__ void wait_async6() {
  asm volatile("s_wait_asynccnt 0x6" ::: "memory");
}
__device__ __forceinline__ void wait_async0() {
  asm volatile("s_wait_asynccnt 0x0" ::: "memory");
}

// ---------------------------------------------------------------------------
// Prep: split fp32 -> (hi, lo) bf16 pair.  x = hi + lo + O(2^-16 * x)
// ---------------------------------------------------------------------------
__global__ __launch_bounds__(256) void split_bf16_kernel(
    const float* __restrict__ src, u16* __restrict__ hi, u16* __restrict__ lo, int n)
{
  int i = blockIdx.x * 256 + threadIdx.x;
  if (i < n) {
    float x = src[i];
    u16 h = f2bf(x);
    hi[i] = h;
    lo[i] = f2bf(x - bf2f(h));
  }
}

// ---------------------------------------------------------------------------
// Shared 128x64 GEMM core (K = 768, bf16-split triple-WMMA, async double buf).
// A: 128 rows (M), B: 64 rows (N); both row-major with HID bf16 per row.
// Wave w (of 8) accumulates rows [16w,16w+16) x 64 N into acc[4] (v8f each).
// ---------------------------------------------------------------------------
__device__ __forceinline__ void stage_tiles(
    const u16* __restrict__ Ah_g, const u16* __restrict__ Al_g,
    const u16* __restrict__ Bh_g, const u16* __restrict__ Bl_g,
    u32 (*AhL)[20], u32 (*AlL)[20], u32 (*BhL)[20], u32 (*BlL)[20],
    int tid, int m0, int n0, int k0)
{
  // A tile: 128 rows x 4 16B-segments (hi & lo): 2 segs per thread each.
  #pragma unroll
  for (int i = 0; i < 2; ++i) {
    const int s = tid + (i << 8);
    const int row = s >> 2, c = s & 3;
    const size_t goff = (size_t)(m0 + row) * HID + k0 + c * 8;
    async_b128((u32)(size_t)&AhL[row][c * 4], Ah_g + goff);
    async_b128((u32)(size_t)&AlL[row][c * 4], Al_g + goff);
  }
  // B tile: 64 rows x 4 segments: 1 seg per thread (hi & lo).
  {
    const int row = tid >> 2, c = tid & 3;
    const size_t goff = (size_t)(n0 + row) * HID + k0 + c * 8;
    async_b128((u32)(size_t)&BhL[row][c * 4], Bh_g + goff);
    async_b128((u32)(size_t)&BlL[row][c * 4], Bl_g + goff);
  }
}

__device__ __forceinline__ void gemm_128x64(
    const u16* __restrict__ Ah_g, const u16* __restrict__ Al_g,
    const u16* __restrict__ Bh_g, const u16* __restrict__ Bl_g,
    int m0, int n0, int tid, v8f acc[4])
{
  __shared__ __align__(16) u32 AhL[2][128][20];   // 80B rows: 16B aligned, no bank conflicts
  __shared__ __align__(16) u32 AlL[2][128][20];
  __shared__ __align__(16) u32 BhL[2][64][20];
  __shared__ __align__(16) u32 BlL[2][64][20];

  const int lane = tid & 31;
  const int wave = tid >> 5;
  const int arow = lane & 15;
  const int g = lane >> 4;                        // K-group select

  int cur = 0;
  stage_tiles(Ah_g, Al_g, Bh_g, Bl_g, AhL[0], AlL[0], BhL[0], BlL[0], tid, m0, n0, 0);

  for (int ki = 0; ki < KCHUNKS; ++ki) {
    const int nxt = cur ^ 1;
    if (ki + 1 < KCHUNKS) {
      stage_tiles(Ah_g, Al_g, Bh_g, Bl_g, AhL[nxt], AlL[nxt], BhL[nxt], BlL[nxt],
                  tid, m0, n0, 32 * (ki + 1));
      wait_async6();   // oldest 6 (this chunk's tiles) retired; next 6 in flight
    } else {
      wait_async0();
    }
    __syncthreads();

    // A operand: bf16 16x32, lane holds K in [8g,8g+8) U [16+8g,16+8g+8)
    const u32* xh = AhL[cur][(wave << 4) + arow];
    const u32* xl = AlL[cur][(wave << 4) + arow];
    const v16bf Ah = mk16(*(const uint4*)(xh + 4 * g), *(const uint4*)(xh + 8 + 4 * g));
    const v16bf Al = mk16(*(const uint4*)(xl + 4 * g), *(const uint4*)(xl + 8 + 4 * g));
    #pragma unroll
    for (int nt = 0; nt < 4; ++nt) {
      // B operand: bf16 32x16, lane (col) holds K in [8g,8g+8) U [16+8g,16+8g+8)
      const u32* wh = BhL[cur][(nt << 4) + arow];
      const u32* wl = BlL[cur][(nt << 4) + arow];
      const v16bf Bh = mk16(*(const uint4*)(wh + 4 * g), *(const uint4*)(wh + 8 + 4 * g));
      const v16bf Bl = mk16(*(const uint4*)(wl + 4 * g), *(const uint4*)(wl + 8 + 4 * g));
      acc[nt] = wmma32bf(Ah, Bh, acc[nt]);   // hi*hi
      acc[nt] = wmma32bf(Al, Bh, acc[nt]);   // lo*hi
      acc[nt] = wmma32bf(Ah, Bl, acc[nt]);   // hi*lo
    }
    __syncthreads();
    cur = nxt;
  }
}

// ---------------------------------------------------------------------------
// Kernel 1: QKV projection + RoPE, scatter to (B, nh, S, 64) fp32.
// Grid: (NQKV/64, MTOT/128), 256 threads. Each WG's 64-wide N slab is exactly
// one (tensor t, head) block, so d and d+32 live in acc[nt] / acc[nt+2].
// ---------------------------------------------------------------------------
__global__ __launch_bounds__(256) void qkv_rope_kernel(
    const u16* __restrict__ Xh, const u16* __restrict__ Xl,
    const u16* __restrict__ Wh, const u16* __restrict__ Wl,
    const int* __restrict__ pos,
    float* __restrict__ qws, float* __restrict__ kws, float* __restrict__ vws)
{
  const int tid = threadIdx.x, lane = tid & 31, wave = tid >> 5;
  const int m0 = blockIdx.y * 128;
  const int n0 = blockIdx.x * 64;
  const int arow = lane & 15;

  v8f acc[4] = {};
  gemm_128x64(Xh, Xl, Wh, Wl, m0, n0, tid, acc);

  const int t  = n0 / HID;                 // 0=Q, 1=K, 2=V
  const int hb = (n0 - t * HID) / HD;
  #pragma unroll
  for (int r = 0; r < 8; ++r) {
    const int m = m0 + (wave << 4) + ((lane >> 4) << 3) + r;
    const int b = m >> 11;                 // SEQ == 2048
    const int s = m & (SEQ - 1);
    const size_t base = ((size_t)(b * NH + hb) * SEQ + s) * HD;
    if (t == 2) {
      #pragma unroll
      for (int nt = 0; nt < 4; ++nt)
        vws[base + (nt << 4) + arow] = acc[nt][r];
    } else {
      float* dst = (t == 0) ? qws : kws;
      const float p = (float)pos[m];
      #pragma unroll
      for (int nt = 0; nt < 2; ++nt) {
        const int d1 = (nt << 4) + arow;                    // 0..31
        const float inv = expf(-0.28782313662425576f * (float)d1); // 1e4^(-d1/32)
        float sn, cs;
        sincosf(p * inv, &sn, &cs);
        const float vlo = acc[nt][r], vhi = acc[nt + 2][r];
        dst[base + d1]      = vlo * cs - vhi * sn;
        dst[base + d1 + 32] = vhi * cs + vlo * sn;
      }
    }
  }
}

// ---------------------------------------------------------------------------
// Kernel 2: sliding-window attention (exact fp32 WMMA path).
// Grid: (SEQ/128, B*NH), 256 threads; wave w owns query tile q0.
// Writes ctx split to bf16 hi/lo for the output GEMM.
// ---------------------------------------------------------------------------
__global__ __launch_bounds__(256) void attn_kernel(
    const float* __restrict__ qws, const float* __restrict__ kws,
    const float* __restrict__ vws,
    u16* __restrict__ ctxh, u16* __restrict__ ctxl)
{
  __shared__ float Ss[8][16][148];  // per-wave scores; stride 148 -> conflict-free
  const int tid = threadIdx.x, lane = tid & 31, wave = tid >> 5;
  const int bh = blockIdx.y;
  const int b = bh / NH, h = bh % NH;
  const int q0 = (blockIdx.x * 8 + wave) * 16;
  const size_t hbase = (size_t)bh * SEQ * HD;
  const float* Qg = qws + hbase;
  const float* Kg = kws + hbase;
  const float* Vg = vws + hbase;
  const int arow = lane & 15;
  const int koff = (lane >> 4) << 1;

  // scores = (Q K^T)/8, masked, into LDS
  for (int kt = 0; kt < 9; ++kt) {
    const int kbase = q0 - HALF_WIN + kt * 16;
    const int kcl = clampi(kbase + arow, 0, SEQ - 1);
    v8f acc = {};
    #pragma unroll
    for (int d0 = 0; d0 < HD; d0 += 4) {
      v2f a  = *(const v2f*)(Qg + (size_t)(q0 + arow) * HD + d0 + koff);
      v2f bb = *(const v2f*)(Kg + (size_t)kcl * HD + d0 + koff);
      acc = wmma4(a, bb, acc);
    }
    #pragma unroll
    for (int r = 0; r < 8; ++r) {
      const int qrow = ((lane >> 4) << 3) + r;
      const int kk = kbase + arow;
      int dd = (q0 + qrow) - kk; dd = dd < 0 ? -dd : dd;
      const bool valid = (kk >= 0) && (kk < SEQ) && (dd <= HALF_WIN);
      Ss[wave][qrow][kt * 16 + arow] = valid ? acc[r] * 0.125f : NEGBIG;
    }
  }

  // exact softmax, one query row per lane (lanes 0..15)
  if (lane < 16) {
    float mx = NEGBIG;
    for (int j = 0; j < 144; ++j) mx = fmaxf(mx, Ss[wave][lane][j]);
    float sum = 0.f;
    for (int j = 0; j < 144; ++j) {
      const float e = expf(Ss[wave][lane][j] - mx);   // masked -> exactly 0
      Ss[wave][lane][j] = e;
      sum += e;
    }
    const float inv = 1.0f / sum;
    for (int j = 0; j < 144; ++j) Ss[wave][lane][j] *= inv;
  }

  // context = P @ V
  v8f out[4] = {};
  for (int kt = 0; kt < 9; ++kt) {
    const int kbase = q0 - HALF_WIN + kt * 16;
    #pragma unroll
    for (int kq = 0; kq < 16; kq += 4) {
      v2f a = *(const v2f*)&Ss[wave][arow][kt * 16 + kq + koff];
      const int kr0 = clampi(kbase + kq + koff,     0, SEQ - 1);
      const int kr1 = clampi(kbase + kq + koff + 1, 0, SEQ - 1);
      #pragma unroll
      for (int nt = 0; nt < 4; ++nt) {
        const int d = (nt << 4) + arow;
        v2f bb;
        bb.x = Vg[(size_t)kr0 * HD + d];   // clamped rows contribute 0 (p==0)
        bb.y = Vg[(size_t)kr1 * HD + d];
        out[nt] = wmma4(a, bb, out[nt]);
      }
    }
  }

  // write ctx rows (b, s, h*64+d) as bf16 hi/lo
  #pragma unroll
  for (int r = 0; r < 8; ++r) {
    const int qrow = ((lane >> 4) << 3) + r;
    const size_t row = ((size_t)b * SEQ + q0 + qrow) * HID + h * HD;
    #pragma unroll
    for (int nt = 0; nt < 4; ++nt) {
      const float v = out[nt][r];
      const u16 hh = f2bf(v);
      ctxh[row + (nt << 4) + arow] = hh;
      ctxl[row + (nt << 4) + arow] = f2bf(v - bf2f(hh));
    }
  }
}

// ---------------------------------------------------------------------------
// Kernel 3: output projection (same bf16-split GEMM core), writes fp32 out.
// Grid: (HID/64, MTOT/128), 256 threads.
// ---------------------------------------------------------------------------
__global__ __launch_bounds__(256) void out_proj_kernel(
    const u16* __restrict__ Ch, const u16* __restrict__ Cl,
    const u16* __restrict__ Wh, const u16* __restrict__ Wl,
    float* __restrict__ out)
{
  const int tid = threadIdx.x, lane = tid & 31, wave = tid >> 5;
  const int m0 = blockIdx.y * 128;
  const int n0 = blockIdx.x * 64;
  const int arow = lane & 15;

  v8f acc[4] = {};
  gemm_128x64(Ch, Cl, Wh, Wl, m0, n0, tid, acc);

  #pragma unroll
  for (int r = 0; r < 8; ++r) {
    const int m = m0 + (wave << 4) + ((lane >> 4) << 3) + r;
    #pragma unroll
    for (int nt = 0; nt < 4; ++nt)
      out[(size_t)m * HID + n0 + (nt << 4) + arow] = acc[nt][r];
  }
}

// ---------------------------------------------------------------------------
extern "C" void kernel_launch(void* const* d_in, const int* in_sizes, int n_in,
                              void* d_out, int out_size, void* d_ws, size_t ws_size,
                              hipStream_t stream) {
  (void)in_sizes; (void)n_in; (void)out_size; (void)ws_size;
  const float* X   = (const float*)d_in[0];   // (B,S,H) fp32
  const int*   pos = (const int*)d_in[1];     // (B,S)
  const float* Wq  = (const float*)d_in[2];   // (3H,H) fp32
  const float* Wo  = (const float*)d_in[3];   // (H,H) fp32
  float* out = (float*)d_out;                 // (B,S,H) fp32

  const int nX  = MTOT * HID;                 // 6291456
  const int nWq = NQKV * HID;                 // 1769472
  const int nWo = HID * HID;                  // 589824
  const size_t perT = (size_t)BSZ * NH * SEQ * HD;  // q/k/v fp32 elements

  // Workspace partition (256B aligned): ~135 MB total.
  char* p = (char*)d_ws;
  auto take = [&p](size_t bytes) {
    char* r = p;
    p += (bytes + 255) & ~(size_t)255;
    return r;
  };
  u16* Xh  = (u16*)take((size_t)nX * 2);
  u16* Xl  = (u16*)take((size_t)nX * 2);
  u16* Wqh = (u16*)take((size_t)nWq * 2);
  u16* Wql = (u16*)take((size_t)nWq * 2);
  u16* Woh = (u16*)take((size_t)nWo * 2);
  u16* Wol = (u16*)take((size_t)nWo * 2);
  float* qws = (float*)take(perT * 4);
  float* kws = (float*)take(perT * 4);
  float* vws = (float*)take(perT * 4);
  u16* ctxh = (u16*)take((size_t)nX * 2);
  u16* ctxl = (u16*)take((size_t)nX * 2);

  dim3 blk(256);
  split_bf16_kernel<<<(nX  + 255) / 256, blk, 0, stream>>>(X,  Xh,  Xl,  nX);
  split_bf16_kernel<<<(nWq + 255) / 256, blk, 0, stream>>>(Wq, Wqh, Wql, nWq);
  split_bf16_kernel<<<(nWo + 255) / 256, blk, 0, stream>>>(Wo, Woh, Wol, nWo);

  qkv_rope_kernel<<<dim3(NQKV / 64, MTOT / 128), blk, 0, stream>>>(
      Xh, Xl, Wqh, Wql, pos, qws, kws, vws);
  attn_kernel<<<dim3(SEQ / 128, BSZ * NH), blk, 0, stream>>>(
      qws, kws, vws, ctxh, ctxl);
  out_proj_kernel<<<dim3(HID / 64, MTOT / 128), blk, 0, stream>>>(
      ctxh, ctxl, Woh, Wol, out);
}
